// AttentionBlock_58445914964210
// MI455X (gfx1250) — matrix operand order repaired and checked by reading, not compiled, not run on hardware
//
#include <hip/hip_runtime.h>
#include <hip/hip_bf16.h>

// ---------------------------------------------------------------------------
// Problem constants (reference: B=2, S=2048, D=1024, H=16, HD=64)
// ---------------------------------------------------------------------------
#define BATCH 2
#define SLEN  2048
#define DDIM  1024
#define HNUM  16
#define HDIM  64
#define QKVN  (HNUM * 3 * HDIM)   // 3072
#define ROWS  (BATCH * SLEN)      // 4096

typedef _Float16 h16;
typedef __attribute__((ext_vector_type(8)))  _Float16 v8h;
typedef __attribute__((ext_vector_type(16))) _Float16 v16h;
typedef __attribute__((ext_vector_type(8)))  float    v8f;

static __device__ __forceinline__ v16h cat16(v8h lo, v8h hi) {
    return __builtin_shufflevector(lo, hi, 0,1,2,3,4,5,6,7,8,9,10,11,12,13,14,15);
}

// ---------------------------------------------------------------------------
// CDNA5 async global->LDS copies (ASYNCcnt path), guarded so compile is safe
// Builtin signature (from hipcc diagnostic): param0 = AS1 (global) int4*,
// param1 = AS3 (LDS) int4*, then imm offset, imm cpol.
// ---------------------------------------------------------------------------
#if __has_builtin(__builtin_amdgcn_global_load_async_to_lds_b128)
#define USE_ASYNC_LDS 1
typedef int v4i_ __attribute__((vector_size(16)));
typedef __attribute__((address_space(1))) v4i_* gbl_v4p;
typedef __attribute__((address_space(3))) v4i_* lds_v4p;
#define ASYNC_CP16(gsrc, ldst)                                                  \
    __builtin_amdgcn_global_load_async_to_lds_b128(                             \
        (gbl_v4p)(void*)(gsrc), (lds_v4p)(void*)(ldst), 0, 0)
static __device__ __forceinline__ void async_wait0() {
#if __has_builtin(__builtin_amdgcn_s_wait_asynccnt)
    __builtin_amdgcn_s_wait_asynccnt(0);
#else
    asm volatile("s_wait_asynccnt 0" ::: "memory");
#endif
}
#else
#define USE_ASYNC_LDS 0
#endif

// Copy 16 halves (32B) global -> LDS
static __device__ __forceinline__ void cp16h(const h16* g, h16* l) {
#if USE_ASYNC_LDS
    ASYNC_CP16(g, l);
    ASYNC_CP16(g + 8, l + 8);
#else
    v8h a0 = *(const v8h*)g, a1 = *(const v8h*)(g + 8);
    *(v8h*)l = a0; *(v8h*)(l + 8) = a1;
#endif
}

static __device__ __forceinline__ void tile_loads_done() {
#if USE_ASYNC_LDS
    async_wait0();
#endif
    __syncthreads();
}

// ---------------------------------------------------------------------------
// LayerNorm over D=1024, one block (256 thr) per row, f16 output
// ---------------------------------------------------------------------------
__global__ __launch_bounds__(256)
void ln_kernel(const float* __restrict__ x, const float* __restrict__ g,
               const float* __restrict__ bt, h16* __restrict__ xnh)
{
    __shared__ float red[32];
    const int r = blockIdx.x;
    const float* xr = x + (size_t)r * DDIM;
    float vals[4], s = 0.f, ss = 0.f;
#pragma unroll
    for (int i = 0; i < 4; ++i) {
        float v = xr[threadIdx.x + i * 256];
        vals[i] = v; s += v; ss += v * v;
    }
#pragma unroll
    for (int m = 16; m >= 1; m >>= 1) {
        s  += __shfl_xor(s,  m, 32);
        ss += __shfl_xor(ss, m, 32);
    }
    const int wave = threadIdx.x >> 5;
    if ((threadIdx.x & 31) == 0) { red[wave] = s; red[8 + wave] = ss; }
    __syncthreads();
    if (threadIdx.x < 32) {
        float a = (threadIdx.x < 8) ? red[threadIdx.x]     : 0.f;
        float b = (threadIdx.x < 8) ? red[8 + threadIdx.x] : 0.f;
#pragma unroll
        for (int m = 4; m >= 1; m >>= 1) {
            a += __shfl_xor(a, m, 32);
            b += __shfl_xor(b, m, 32);
        }
        if (threadIdx.x == 0) { red[16] = a; red[17] = b; }
    }
    __syncthreads();
    const float mu   = red[16] * (1.f / DDIM);
    const float var  = red[17] * (1.f / DDIM) - mu * mu;
    const float rstd = rsqrtf(var + 1e-6f);
#pragma unroll
    for (int i = 0; i < 4; ++i) {
        int c = threadIdx.x + i * 256;
        float xn = (vals[i] - mu) * rstd * g[c] + bt[c];
        xnh[(size_t)r * DDIM + c] = (h16)xn;
    }
}

// ---------------------------------------------------------------------------
// f32 [K][N] -> f16 transposed [N][K]  (one-time; weights reused by all tiles)
// ---------------------------------------------------------------------------
__global__ __launch_bounds__(256)
void cvtT_kernel(const float* __restrict__ in, h16* __restrict__ out,
                 int Kd, int Nd)
{
    int idx = blockIdx.x * 256 + threadIdx.x;
    if (idx >= Kd * Nd) return;
    int n = idx / Kd, k = idx - n * Kd;
    out[idx] = (h16)in[(size_t)k * Nd + n];
}

// ---------------------------------------------------------------------------
// Tiled WMMA GEMM: C[M,N] = A[M,K] * W[K,N], with W supplied transposed
// BT[N][K] (f16). Block tile 128x128, 8 waves (4x2), wave tile 32x64
// (2x4 wmma tiles), K-step 32. Both LDS tiles are straight row copies.
// EPI==0: QKV epilogue (+b_qkv, scatter to Q(scaled)/K/V^T per-head buffers)
// EPI==1: out-proj epilogue (+b_out, f32 store)
// ---------------------------------------------------------------------------
template<int EPI>
__global__ __launch_bounds__(256)
void gemm_f16(const h16* __restrict__ A, const h16* __restrict__ BT,
              const float* __restrict__ bias,
              float* __restrict__ outF,
              h16* __restrict__ Qh, h16* __restrict__ Kh, h16* __restrict__ Vt,
              int M, int N, int K)
{
    __shared__ alignas(16) h16 As[128][40];   // [m][k] + pad
    __shared__ alignas(16) h16 Bs[128][40];   // [n][k] + pad

    const int tid    = threadIdx.x;
    const int lane   = tid & 31;
    const int wave   = tid >> 5;
    const int waveM  = wave & 3;     // 4 x 32 rows
    const int waveN  = wave >> 2;    // 2 x 64 cols
    const int n16    = lane & 15;
    const int halfId = lane >> 4;

    const int mBase = blockIdx.y * 128;
    const int nBase = blockIdx.x * 128;

    v8f acc[2][4] = {};

    const int lRow = tid >> 1;          // 0..127
    const int lSeg = (tid & 1) * 16;    // 0 / 16

    for (int k0 = 0; k0 < K; k0 += 32) {
        __syncthreads();
        const h16* ga = A  + (size_t)(mBase + lRow) * K + k0 + lSeg;
        const h16* gb = BT + (size_t)(nBase + lRow) * K + k0 + lSeg;
        cp16h(ga, &As[lRow][lSeg]);
        cp16h(gb, &Bs[lRow][lSeg]);
        if (k0 + 32 < K) {               // prefetch next k-step
            __builtin_prefetch(ga + 32, 0, 1);
            __builtin_prefetch(gb + 32, 0, 1);
        }
        tile_loads_done();

        v16h aF[2], bF[4];
#pragma unroll
        for (int ms = 0; ms < 2; ++ms) {
            const h16* p = &As[waveM * 32 + ms * 16 + n16][halfId * 16];
            aF[ms] = cat16(*(const v8h*)p, *(const v8h*)(p + 8));
        }
#pragma unroll
        for (int ns = 0; ns < 4; ++ns) {
            const h16* p = &Bs[waveN * 64 + ns * 16 + n16][halfId * 16];
            bF[ns] = cat16(*(const v8h*)p, *(const v8h*)(p + 8));
        }
#pragma unroll
        for (int ms = 0; ms < 2; ++ms)
#pragma unroll
            for (int ns = 0; ns < 4; ++ns)
                acc[ms][ns] = __builtin_amdgcn_wmma_f32_16x16x32_f16(
                    false, aF[ms], false, bF[ns], (short)0, acc[ms][ns], false, false);
    }

    // Epilogue
#pragma unroll
    for (int ms = 0; ms < 2; ++ms) {
#pragma unroll
        for (int ns = 0; ns < 4; ++ns) {
            const int c = nBase + waveN * 64 + ns * 16 + n16;
#pragma unroll
            for (int j = 0; j < 8; ++j) {
                const int r = mBase + waveM * 32 + ms * 16 + j + 8 * halfId;
                float v = acc[ms][ns][j] + bias[c];
                if (EPI == 0) {
                    const int b = r >> 11, s = r & (SLEN - 1);
                    const int h = c / 192, f = c - h * 192;
                    const size_t bh = (size_t)(b * HNUM + h);
                    if (f < 64)
                        Qh[(bh * SLEN + s) * HDIM + f] = (h16)(v * 0.125f); // HD^-0.5
                    else if (f < 128)
                        Kh[(bh * SLEN + s) * HDIM + (f - 64)] = (h16)v;
                    else
                        Vt[(bh * HDIM + (f - 128)) * SLEN + s] = (h16)v;    // transposed V
                } else {
                    outF[(size_t)r * N + c] = v;
                }
            }
        }
    }
}

// ---------------------------------------------------------------------------
// Flash attention: block = (q-tile of 64, head, batch), 128 threads (4 waves),
// each wave owns 16 query rows; causal key tiles of 64. QK^T and PV via WMMA.
// Row-sum l is accumulated through the matmul itself (ones-column fragment),
// so only the row-max needs a cross-lane butterfly.
// ---------------------------------------------------------------------------
__global__ __launch_bounds__(128)
void attn_kernel(const h16* __restrict__ Qh, const h16* __restrict__ Kh,
                 const h16* __restrict__ Vt, h16* __restrict__ Oh)
{
    __shared__ alignas(16) h16 Ks[64][72];       // [key][hd] + pad
    __shared__ alignas(16) h16 Vts[64][72];      // [hd][key] + pad
    __shared__ alignas(16) h16 Ps[4][16][72];    // per-wave P tile [row][key]

    const int tid    = threadIdx.x;
    const int lane   = tid & 31;
    const int wave   = tid >> 5;
    const int n16    = lane & 15;
    const int halfId = lane >> 4;

    const int qblk = blockIdx.x;
    const int h    = blockIdx.y;
    const int b    = blockIdx.z;

    const size_t bh = (size_t)(b * HNUM + h);
    const h16* Qg = Qh + bh * SLEN * HDIM;
    const h16* Kg = Kh + bh * SLEN * HDIM;
    const h16* Vg = Vt + bh * HDIM * SLEN;

    const int qwb = qblk * 64 + wave * 16;

    // Q fragments for the two hd 32-chunks (Q already scaled)
    v16h qF[2];
    {
        const h16* qr = Qg + (size_t)(qwb + n16) * HDIM;
#pragma unroll
        for (int s = 0; s < 2; ++s) {
            const h16* p = qr + s * 32 + halfId * 16;
            qF[s] = cat16(*(const v8h*)p, *(const v8h*)(p + 8));
        }
    }

    // "ones" B fragment: column 0 = 1, used to accumulate row-sums via WMMA
    v16h onesF;
    {
        const h16 o = (n16 == 0) ? (h16)1.f : (h16)0.f;
#pragma unroll
        for (int i = 0; i < 16; ++i) onesF[i] = o;
    }

    v8f accO[4] = {};
    v8f accL = {};
    float mrow[8];
#pragma unroll
    for (int j = 0; j < 8; ++j) mrow[j] = -3.0e38f;

    const int lRow = tid >> 1;          // 0..63
    const int lSeg = (tid & 1) * 32;    // 0 / 32

    const int nkt = qblk + 1;           // 64-key tiles covering causal range
    for (int kt = 0; kt < nkt; ++kt) {
        const int kb = kt * 64;
        __syncthreads();
        { // K tile: 64 keys x 64 hd ; V^T tile: 64 hd x 64 keys
            const h16* gk = Kg + (size_t)(kb + lRow) * HDIM + lSeg;
            const h16* gv = Vg + (size_t)lRow * SLEN + kb + lSeg;
            cp16h(gk,      &Ks[lRow][lSeg]);
            cp16h(gk + 16, &Ks[lRow][lSeg + 16]);
            cp16h(gv,      &Vts[lRow][lSeg]);
            cp16h(gv + 16, &Vts[lRow][lSeg + 16]);
        }
        tile_loads_done();

        // S = Q * K^T : four 16-key subtiles, K-dim = 64 (2 wmma each)
        v8f sc[4];
#pragma unroll
        for (int ns = 0; ns < 4; ++ns) {
            v8f c = {};
#pragma unroll
            for (int s = 0; s < 2; ++s) {
                const h16* p = &Ks[ns * 16 + n16][s * 32 + halfId * 16];
                v16h bF = cat16(*(const v8h*)p, *(const v8h*)(p + 8));
                c = __builtin_amdgcn_wmma_f32_16x16x32_f16(
                        false, qF[s], false, bF, (short)0, c, false, false);
            }
            sc[ns] = c;
        }

        // causal mask + online softmax (rows live in 16-lane halves)
#pragma unroll
        for (int j = 0; j < 8; ++j) {
            const int q = qwb + j + 8 * halfId;
            float pv[4];
            float t = -3.0e38f;
#pragma unroll
            for (int ns = 0; ns < 4; ++ns) {
                const int col = kb + ns * 16 + n16;
                const float sv = (col <= q) ? sc[ns][j] : -3.0e38f;
                pv[ns] = sv;
                t = fmaxf(t, sv);
            }
#pragma unroll
            for (int m = 8; m >= 1; m >>= 1) t = fmaxf(t, __shfl_xor(t, m, 32));
            const float mnew = fmaxf(mrow[j], t);
            const float sf = __expf(mrow[j] - mnew);
            mrow[j] = mnew;
#pragma unroll
            for (int t4 = 0; t4 < 4; ++t4) accO[t4][j] *= sf;
            accL[j] *= sf;
#pragma unroll
            for (int ns = 0; ns < 4; ++ns)
                Ps[wave][j + 8 * halfId][ns * 16 + n16] = (h16)__expf(pv[ns] - mnew);
        }
        __syncthreads();

        // O += P(16x64) * V(64x64) ; l += P * ones (one extra wmma per chunk)
#pragma unroll
        for (int c = 0; c < 2; ++c) {
            const h16* pp = &Ps[wave][n16][c * 32 + halfId * 16];
            v16h pF = cat16(*(const v8h*)pp, *(const v8h*)(pp + 8));
#pragma unroll
            for (int t4 = 0; t4 < 4; ++t4) {
                const h16* vp = &Vts[t4 * 16 + n16][c * 32 + halfId * 16];
                v16h bF = cat16(*(const v8h*)vp, *(const v8h*)(vp + 8));
                accO[t4] = __builtin_amdgcn_wmma_f32_16x16x32_f16(
                               false, pF, false, bF, (short)0, accO[t4], false, false);
            }
            accL = __builtin_amdgcn_wmma_f32_16x16x32_f16(
                       false, pF, false, onesF, (short)0, accL, false, false);
        }
    }

    // broadcast l (lives in column 0 => lanes 0 and 16), normalize, store f16
    float lrow[8];
#pragma unroll
    for (int j = 0; j < 8; ++j) lrow[j] = __shfl(accL[j], halfId * 16, 32);
#pragma unroll
    for (int t4 = 0; t4 < 4; ++t4) {
#pragma unroll
        for (int j = 0; j < 8; ++j) {
            const int q = qwb + j + 8 * halfId;
            const float o = accO[t4][j] / lrow[j];
            Oh[((size_t)b * SLEN + q) * DDIM + h * HDIM + t4 * 16 + n16] = (h16)o;
        }
    }
}

// ---------------------------------------------------------------------------
// Launch
// ---------------------------------------------------------------------------
extern "C" void kernel_launch(void* const* d_in, const int* in_sizes, int n_in,
                              void* d_out, int out_size, void* d_ws, size_t ws_size,
                              hipStream_t stream)
{
    const float* x      = (const float*)d_in[0];
    const float* ln_s   = (const float*)d_in[1];
    const float* ln_b   = (const float*)d_in[2];
    const float* w_qkv  = (const float*)d_in[3];
    const float* b_qkv  = (const float*)d_in[4];
    const float* w_out  = (const float*)d_in[5];
    const float* b_out  = (const float*)d_in[6];
    // d_in[7] (mask) unused: causal mask computed analytically
    (void)in_sizes; (void)n_in; (void)out_size; (void)ws_size;

    char* ws = (char*)d_ws;
    h16* xnh   = (h16*)(ws);                          // 8 MB  [r][k]
    h16* wqkvT = (h16*)(ws + (8u << 20));             // 6 MB  [3072][1024]
    h16* woutT = (h16*)(ws + (14u << 20));            // 2 MB  [1024][1024]
    h16* Qh    = (h16*)(ws + (16u << 20));            // 8 MB  [b,h,s,hd] scaled
    h16* Kh    = (h16*)(ws + (24u << 20));            // 8 MB  [b,h,s,hd]
    h16* Vth   = (h16*)(ws + (32u << 20));            // 8 MB  [b,h,hd,s]
    h16* Oh    = (h16*)(ws + (40u << 20));            // 8 MB  [b,s,h*hd]
    float* out = (float*)d_out;

    ln_kernel<<<ROWS, 256, 0, stream>>>(x, ln_s, ln_b, xnh);

    cvtT_kernel<<<(DDIM * QKVN + 255) / 256, 256, 0, stream>>>(w_qkv, wqkvT, DDIM, QKVN);
    cvtT_kernel<<<(DDIM * DDIM + 255) / 256, 256, 0, stream>>>(w_out, woutT, DDIM, DDIM);

    dim3 g1(QKVN / 128, ROWS / 128);   // 24 x 32
    gemm_f16<0><<<g1, 256, 0, stream>>>(xnh, wqkvT, b_qkv, nullptr,
                                        Qh, Kh, Vth, ROWS, QKVN, DDIM);

    attn_kernel<<<dim3(SLEN / 64, HNUM, BATCH), 128, 0, stream>>>(Qh, Kh, Vth, Oh);

    dim3 g2(DDIM / 128, ROWS / 128);   // 8 x 32
    gemm_f16<1><<<g2, 256, 0, stream>>>(Oh, woutT, b_out, out,
                                        nullptr, nullptr, nullptr, ROWS, DDIM, DDIM);
}
